// Attention_2508260901645
// MI455X (gfx1250) — matrix operand order repaired and checked by reading, not compiled, not run on hardware
//
#include <hip/hip_runtime.h>
#include <hip/hip_bf16.h>

// ---------------------------------------------------------------------------
// CDNA5 (gfx1250) causal multi-head attention, bf16 WMMA (16x16x32), fp32 accum
// + GLOBAL_LOAD_ASYNC_TO_LDS (ASYNCcnt) and TDM tensor_load_to_lds (TENSORcnt)
// ---------------------------------------------------------------------------

typedef __bf16 bf16;
typedef bf16  bf16x8 __attribute__((ext_vector_type(8)));
typedef bf16  v16bf  __attribute__((ext_vector_type(16)));
typedef float v8f    __attribute__((ext_vector_type(8)));
typedef unsigned int u32x4 __attribute__((ext_vector_type(4)));
typedef int          i32x8 __attribute__((ext_vector_type(8)));
typedef int          i32x4 __attribute__((ext_vector_type(4)));

#define EMB   1024
#define NE3   3072
#define SEQ   2048
#define NHEAD 16
#define HD    64
#define NB    2

#define HAVE_TDM __has_builtin(__builtin_amdgcn_tensor_load_to_lds)

// Build a 16-element bf16 A/B fragment from two 16-byte chunks.
__device__ __forceinline__ v16bf make_frag(const bf16* p0, const bf16* p1) {
    bf16x8 lo = *(const bf16x8*)p0;
    bf16x8 hi = *(const bf16x8*)p1;
    v16bf f;
#pragma unroll
    for (int i = 0; i < 8; ++i) { f[i] = lo[i]; f[i + 8] = hi[i]; }
    return f;
}

__device__ __forceinline__ v8f wmma_bf16(v16bf a, v16bf b, v8f c) {
    return __builtin_amdgcn_wmma_f32_16x16x32_bf16(
        /*neg_a=*/false, a, /*neg_b=*/false, b,
        /*c_mod=*/(short)0, c, /*reuse_a=*/false, /*reuse_b=*/false);
}

// 16B/lane global -> LDS async copy (VFLAT GLOBAL_LOAD_ASYNC_TO_LDS_B128).
// Generic pointers to __shared__ hold the LDS byte address in their low 32
// bits (ISA 10.2 aperture rules), which is what the VDST operand wants.
__device__ __forceinline__ void async_copy_b128(void* lds_dst, const void* gsrc) {
    unsigned lds_off = (unsigned)(unsigned long long)lds_dst;
    unsigned long long gaddr = (unsigned long long)gsrc;
    asm volatile("global_load_async_to_lds_b128 %0, %1, off"
                 :: "v"(lds_off), "v"(gaddr) : "memory");
}
__device__ __forceinline__ void wait_async0() {
    asm volatile("s_wait_asynccnt 0x0" ::: "memory");
}
__device__ __forceinline__ void wait_tensor0() {
#if __has_builtin(__builtin_amdgcn_s_wait_tensorcnt)
    __builtin_amdgcn_s_wait_tensorcnt((short)0);
#else
    asm volatile("s_wait_tensorcnt 0x0" ::: "memory");
#endif
}

#if HAVE_TDM
// 2D bf16 tile, global (row_stride elems) -> LDS, optional LDS row padding.
// D# layout per CDNA5 ISA 8.3 (group0) / 8.4 (group1); groups 2/3 unused (2D).
// This toolchain exposes the 6-arg builtin (extra i32x8 group before cpol).
__device__ __forceinline__ void tdm_load_2d_bf16(
    void* lds_base, const void* gbase, int tile_d0, int tile_d1,
    int row_stride_elems, int pad_interval, int pad_amount /* encoded; -1=off */)
{
    const unsigned lds_addr = (unsigned)(unsigned long long)lds_base;
    const unsigned long long ga = (unsigned long long)gbase;
    u32x4 g0;
    g0[0] = 1u;                                             // count=1
    g0[1] = lds_addr;                                       // lds_addr
    g0[2] = (unsigned)ga;                                   // global_addr lo
    g0[3] = (unsigned)((ga >> 32) & 0x01FFFFFFull)          // global_addr hi
          | (2u << 30);                                     // type=2 (image)
    unsigned w0 = (1u << 16);                               // data_size=1 (2B)
    if (pad_amount >= 0)
        w0 |= (1u << 20) | ((unsigned)pad_interval << 22)
            | ((unsigned)pad_amount << 25);
    i32x8 g1;
    g1[0] = (int)w0;
    g1[1] = (int)((unsigned)tile_d0 << 16);                 // tensor_dim0 lo
    g1[2] = (int)((unsigned)tile_d1 << 16);                 // tensor_dim1 lo
    g1[3] = (int)((unsigned)tile_d0 << 16);                 // tile_dim0
    g1[4] = tile_d1;                                        // tile_dim1
    g1[5] = row_stride_elems;                               // dim0_stride lo
    g1[6] = 0;
    g1[7] = 0;
    const i32x4 z4 = {0, 0, 0, 0};
    const i32x8 z8 = {0, 0, 0, 0, 0, 0, 0, 0};
    __builtin_amdgcn_tensor_load_to_lds(g0, g1, z4, z4, z8, 0);
}
#endif

// ---------------------------------------------------------------------------
// Kernel 1: qkv = x @ w_attn + b_attn, scattered into Q(scaled)/K/V [B,H,S,64]
// Block: 256 threads (8 waves), tile 128(M) x 64(N), K step 32.
// 192 = 3*64, so a 64-wide N-tile maps to exactly one (head, Q/K/V) pair:
// the epilogue routing is block-uniform (no per-lane division / branching).
// ---------------------------------------------------------------------------
__global__ __launch_bounds__(256) void qkv_gemm(
    const float* __restrict__ x, const float* __restrict__ w,
    const float* __restrict__ bias,
    bf16* __restrict__ Qb, bf16* __restrict__ Kb, bf16* __restrict__ Vb)
{
    __shared__ bf16 lx[128][40];   // x tile, row-major [m][k], 80B row stride
    __shared__ bf16 lw[64][40];    // w tile transposed [n][k]

    const int tid  = threadIdx.x;
    const int lane = tid & 31;
    const int wave = tid >> 5;
    const int wm   = wave & 3;     // 4 waves along M (32 rows each)
    const int wn   = wave >> 2;    // 2 waves along N (32 cols each)
    const int m0   = blockIdx.y * 128;
    const int n0   = blockIdx.x * 64;

    v8f acc[2][2] = {};

    const int kbA = (lane >> 4) * 8;    // A fragment K chunk base
    const int kbB = (lane >> 4) * 8;    // B fragment K base
    const int rA0 = wm * 32 + (lane & 15);
    const int cB0 = wn * 32 + (lane & 15);

    const int srow = tid >> 3;          // x staging row (0..31), col group
    const int scol = (tid & 7) * 4;
    const int swn  = tid & 63;          // w staging column
    const int swk  = tid >> 6;          // w staging k quarter

    for (int k0 = 0; k0 < EMB; k0 += 32) {
        // prefetch next K tile into near caches while we convert this one
        if (k0 + 32 < EMB) {
            __builtin_prefetch(x + (m0 + srow) * EMB + (k0 + 32) + scol, 0, 3);
            __builtin_prefetch(w + (k0 + 32 + swk * 8) * NE3 + n0 + swn, 0, 3);
        }
        // stage x tile: 128x32 fp32 -> bf16
#pragma unroll
        for (int rp = 0; rp < 4; ++rp) {
            const int r = srow + rp * 32;
            const float4 v = *(const float4*)(x + (m0 + r) * EMB + k0 + scol);
            lx[r][scol + 0] = (bf16)v.x; lx[r][scol + 1] = (bf16)v.y;
            lx[r][scol + 2] = (bf16)v.z; lx[r][scol + 3] = (bf16)v.w;
        }
        // stage w tile transposed: lw[n][k] = w[k0+k][n0+n]
#pragma unroll
        for (int j = 0; j < 8; ++j) {
            const int k = swk * 8 + j;
            lw[swn][k] = (bf16)w[(k0 + k) * NE3 + n0 + swn];
        }
        __syncthreads();

        v16bf af[2], bf_[2];
#pragma unroll
        for (int i = 0; i < 2; ++i) {
            const int row = rA0 + i * 16;
            af[i] = make_frag(&lx[row][kbA], &lx[row][kbA + 16]);
        }
#pragma unroll
        for (int j = 0; j < 2; ++j) {
            const int col = cB0 + j * 16;
            bf_[j] = make_frag(&lw[col][kbB], &lw[col][kbB + 16]);
        }
#pragma unroll
        for (int i = 0; i < 2; ++i)
#pragma unroll
            for (int j = 0; j < 2; ++j)
                acc[i][j] = wmma_bf16(af[i], bf_[j], acc[i][j]);
        __syncthreads();
    }

    // epilogue: block-uniform head / Q-K-V routing
    const int h  = blockIdx.x / 3;          // scalar
    const int ty = blockIdx.x % 3;          // scalar
    bf16* __restrict__ dst = (ty == 0) ? Qb : (ty == 1) ? Kb : Vb;
    const float scale = (ty == 0) ? 0.125f : 1.0f;   // fold 1/sqrt(64) into Q

    const int hi8 = (lane >> 4) * 8;
    float bj[2];
    int   dvj[2];
#pragma unroll
    for (int j = 0; j < 2; ++j) {
        dvj[j] = wn * 32 + j * 16 + (lane & 15);     // dv in 0..63
        bj[j]  = bias[n0 + dvj[j]];
    }
#pragma unroll
    for (int i = 0; i < 2; ++i)
#pragma unroll
        for (int j = 0; j < 2; ++j)
#pragma unroll
            for (int r = 0; r < 8; ++r) {
                const int m  = m0 + wm * 32 + i * 16 + r + hi8;
                const int bb = m >> 11;              // m / SEQ
                const int s  = m & 2047;
                const float v = (acc[i][j][r] + bj[j]) * scale;
                dst[(((bb * NHEAD + h) * SEQ) + s) * HD + dvj[j]] = (bf16)v;
            }
}

// ---------------------------------------------------------------------------
// Kernel 2: flash attention over one (b,h), 64 q-rows per block (4 waves).
// K tile staged by TDM (one tensor_load_to_lds per tile, wave 0, TENSORcnt,
// D# padding reproducing the 144B LDS row stride); V transposed through VGPRs.
// ---------------------------------------------------------------------------
__global__ __launch_bounds__(128) void attn(
    const bf16* __restrict__ Qb, const bf16* __restrict__ Kb,
    const bf16* __restrict__ Vb, bf16* __restrict__ Ab)
{
    __shared__ bf16 lK[64][72];        // K tile row-major   [kv][d]
    __shared__ bf16 lV[64][72];        // V tile transposed  [d][kv]
    __shared__ bf16 lP[4][16][72];     // per-wave P tile    [qrow][kv]

    const int tid  = threadIdx.x;
    const int lane = tid & 31;
    const int wave = tid >> 5;
    const int bh   = blockIdx.y;       // b*16 + h
    const int qt   = blockIdx.x;       // q tile (64 rows)
    const int qrow_w = qt * 64 + wave * 16;

    const int kb8 = (lane >> 4) * 8;
    const int kbB = (lane >> 4) * 8;
    const int hi8 = (lane >> 4) * 8;

    // Q A-fragments straight from global (rows contiguous in d)
    const bf16* Qp = Qb + (bh * SEQ + qrow_w + (lane & 15)) * HD;
    v16bf qf[2];
    qf[0] = make_frag(Qp + kb8,      Qp + kb8 + 16);
    qf[1] = make_frag(Qp + 32 + kb8, Qp + 32 + kb8 + 16);

    v8f o[4] = {};
    float mrow[8], lrow[8];
#pragma unroll
    for (int r = 0; r < 8; ++r) { mrow[r] = -3.0e38f; lrow[r] = 0.0f; }

    for (int kt = 0; kt <= qt; ++kt) {
        const int kv0 = kt * 64;
#if HAVE_TDM
        // K tile: single TDM DMA, row=128B + 16B LDS pad (interval 32 dw, 4 dw)
        if (wave == 0) {
            tdm_load_2d_bf16(&lK[0][0], Kb + (bh * SEQ + kv0) * HD,
                             HD, 64, HD, /*pad_interval=*/4, /*pad_amount=*/3);
        }
        // V tile transposed through VGPRs (all threads)
#pragma unroll
        for (int pass = 0; pass < 4; ++pass) {
            const int idx = tid + pass * 128;        // 0..511
            const int row = idx >> 3;
            const int c8  = (idx & 7) * 8;
            bf16x8 vv = *(const bf16x8*)(Vb + (bh * SEQ + kv0 + row) * HD + c8);
#pragma unroll
            for (int e = 0; e < 8; ++e) lV[c8 + e][row] = vv[e];
        }
        if (wave == 0) wait_tensor0();
#else
        // fallback: K via async-to-LDS, V transposed through VGPRs
#pragma unroll
        for (int pass = 0; pass < 4; ++pass) {
            const int idx = tid + pass * 128;
            const int row = idx >> 3;
            const int c8  = (idx & 7) * 8;
            const int g   = (bh * SEQ + kv0 + row) * HD + c8;
            async_copy_b128(&lK[row][c8], Kb + g);
            bf16x8 vv = *(const bf16x8*)(Vb + g);
#pragma unroll
            for (int e = 0; e < 8; ++e) lV[c8 + e][row] = vv[e];
        }
        wait_async0();
#endif
        __syncthreads();

        // S = Q K^T (fp32 accum), 4 tiles of 16 kv columns
        v8f sc[4] = {};
#pragma unroll
        for (int j = 0; j < 4; ++j) {
            const int kcol = j * 16 + (lane & 15);
            v16bf b0 = make_frag(&lK[kcol][kbB],      &lK[kcol][kbB + 16]);
            v16bf b1 = make_frag(&lK[kcol][32 + kbB], &lK[kcol][32 + kbB + 16]);
            sc[j] = wmma_bf16(qf[0], b0, sc[j]);
            sc[j] = wmma_bf16(qf[1], b1, sc[j]);
        }

        // causal mask on the diagonal tile (reference uses -1e5)
        if (kt == qt) {
#pragma unroll
            for (int j = 0; j < 4; ++j)
#pragma unroll
                for (int r = 0; r < 8; ++r) {
                    const int rg = qrow_w + r + hi8;
                    const int cg = kv0 + j * 16 + (lane & 15);
                    if (cg > rg) sc[j][r] = -100000.0f;
                }
        }

        // online softmax (row lives in a 16-lane group -> xor 1,2,4,8)
#pragma unroll
        for (int r = 0; r < 8; ++r) {
            float mx = sc[0][r];
#pragma unroll
            for (int j = 1; j < 4; ++j) mx = fmaxf(mx, sc[j][r]);
            mx = fmaxf(mx, __shfl_xor(mx, 1));
            mx = fmaxf(mx, __shfl_xor(mx, 2));
            mx = fmaxf(mx, __shfl_xor(mx, 4));
            mx = fmaxf(mx, __shfl_xor(mx, 8));
            const float mnew  = fmaxf(mrow[r], mx);
            const float alpha = __expf(mrow[r] - mnew);
            float sum = 0.0f;
#pragma unroll
            for (int j = 0; j < 4; ++j) {
                const float p = __expf(sc[j][r] - mnew);
                sc[j][r] = p;
                sum += p;
            }
            sum += __shfl_xor(sum, 1);
            sum += __shfl_xor(sum, 2);
            sum += __shfl_xor(sum, 4);
            sum += __shfl_xor(sum, 8);
            lrow[r] = lrow[r] * alpha + sum;
            mrow[r] = mnew;
#pragma unroll
            for (int j = 0; j < 4; ++j) o[j][r] *= alpha;
            // park P in per-wave LDS to re-swizzle into A-fragment layout
            const int prow = r + hi8;
#pragma unroll
            for (int j = 0; j < 4; ++j)
                lP[wave][prow][j * 16 + (lane & 15)] = (bf16)sc[j][r];
        }

        // O += P @ V
        const int pr = lane & 15;
        v16bf pf0 = make_frag(&lP[wave][pr][kb8],      &lP[wave][pr][kb8 + 16]);
        v16bf pf1 = make_frag(&lP[wave][pr][32 + kb8], &lP[wave][pr][32 + kb8 + 16]);
#pragma unroll
        for (int j = 0; j < 4; ++j) {
            const int dv = j * 16 + (lane & 15);
            v16bf v0 = make_frag(&lV[dv][kbB],      &lV[dv][kbB + 16]);
            v16bf v1 = make_frag(&lV[dv][32 + kbB], &lV[dv][32 + kbB + 16]);
            o[j] = wmma_bf16(pf0, v0, o[j]);
            o[j] = wmma_bf16(pf1, v1, o[j]);
        }
        __syncthreads();
    }

    // normalize and write A [B,S,E] (heads concatenated), bf16
    const int b = bh >> 4, h = bh & 15;
#pragma unroll
    for (int j = 0; j < 4; ++j)
#pragma unroll
        for (int r = 0; r < 8; ++r) {
            const int s  = qrow_w + r + hi8;
            const int dv = j * 16 + (lane & 15);
            const float v = o[j][r] / lrow[r];
            Ab[(b * SEQ + s) * EMB + h * HD + dv] = (bf16)v;
        }
}

// ---------------------------------------------------------------------------
// Kernel 3: out = A @ w_proj + b_proj, fp32 output. Same tiling as kernel 1.
// A tile is already bf16 -> staged with async-to-LDS DMA (ASYNCcnt).
// ---------------------------------------------------------------------------
__global__ __launch_bounds__(256) void proj_gemm(
    const bf16* __restrict__ A, const float* __restrict__ w,
    const float* __restrict__ bias, float* __restrict__ out)
{
    __shared__ bf16 la[128][40];
    __shared__ bf16 lw[64][40];

    const int tid  = threadIdx.x;
    const int lane = tid & 31;
    const int wave = tid >> 5;
    const int wm   = wave & 3;
    const int wn   = wave >> 2;
    const int m0   = blockIdx.y * 128;
    const int n0   = blockIdx.x * 64;

    v8f acc[2][2] = {};
    const int kbA = (lane >> 4) * 8;
    const int kbB = (lane >> 4) * 8;
    const int rA0 = wm * 32 + (lane & 15);
    const int cB0 = wn * 32 + (lane & 15);
    const int swn = tid & 63;
    const int swk = tid >> 6;

    for (int k0 = 0; k0 < EMB; k0 += 32) {
        if (k0 + 32 < EMB)
            __builtin_prefetch(w + (k0 + 32 + swk * 8) * EMB + n0 + swn, 0, 3);
        // stage A tile (already bf16) via async DMA: 128x32 halves, 512 chunks
#pragma unroll
        for (int pass = 0; pass < 2; ++pass) {
            const int idx = tid + pass * 256;
            const int row = idx >> 2;
            const int c8  = (idx & 3) * 8;
            async_copy_b128(&la[row][c8], A + (m0 + row) * EMB + k0 + c8);
        }
        // stage w_proj tile transposed, fp32 -> bf16
#pragma unroll
        for (int j = 0; j < 8; ++j) {
            const int k = swk * 8 + j;
            lw[swn][k] = (bf16)w[(k0 + k) * EMB + n0 + swn];
        }
        wait_async0();
        __syncthreads();

        v16bf af[2], bf_[2];
#pragma unroll
        for (int i = 0; i < 2; ++i) {
            const int row = rA0 + i * 16;
            af[i] = make_frag(&la[row][kbA], &la[row][kbA + 16]);
        }
#pragma unroll
        for (int j = 0; j < 2; ++j) {
            const int col = cB0 + j * 16;
            bf_[j] = make_frag(&lw[col][kbB], &lw[col][kbB + 16]);
        }
#pragma unroll
        for (int i = 0; i < 2; ++i)
#pragma unroll
            for (int j = 0; j < 2; ++j)
                acc[i][j] = wmma_bf16(af[i], bf_[j], acc[i][j]);
        __syncthreads();
    }

    const int hi8 = (lane >> 4) * 8;
    float bj[2];
#pragma unroll
    for (int j = 0; j < 2; ++j)
        bj[j] = bias[n0 + wn * 32 + j * 16 + (lane & 15)];
#pragma unroll
    for (int i = 0; i < 2; ++i)
#pragma unroll
        for (int j = 0; j < 2; ++j)
#pragma unroll
            for (int r = 0; r < 8; ++r) {
                const int m = m0 + wm * 32 + i * 16 + r + hi8;
                const int n = n0 + wn * 32 + j * 16 + (lane & 15);
                out[m * EMB + n] = acc[i][j][r] + bj[j];
            }
}

// ---------------------------------------------------------------------------
extern "C" void kernel_launch(void* const* d_in, const int* in_sizes, int n_in,
                              void* d_out, int out_size, void* d_ws, size_t ws_size,
                              hipStream_t stream) {
    const float* x      = (const float*)d_in[0];  // [2,2048,1024]
    const float* w_attn = (const float*)d_in[1];  // [1024,3072]
    const float* b_attn = (const float*)d_in[2];  // [3072]
    const float* w_proj = (const float*)d_in[3];  // [1024,1024]
    const float* b_proj = (const float*)d_in[4];  // [1024]
    float* out = (float*)d_out;                   // [2,2048,1024]

    const size_t NQKV = (size_t)NB * NHEAD * SEQ * HD;  // 4M elements
    bf16* Qb = (bf16*)d_ws;
    bf16* Kb = Qb + NQKV;
    bf16* Vb = Kb + NQKV;
    bf16* Ab = Vb + NQKV;                               // 32 MB total

    qkv_gemm<<<dim3(NE3 / 64, (NB * SEQ) / 128), 256, 0, stream>>>(
        x, w_attn, b_attn, Qb, Kb, Vb);
    attn<<<dim3(SEQ / 64, NB * NHEAD), 128, 0, stream>>>(Qb, Kb, Vb, Ab);
    proj_gemm<<<dim3(EMB / 64, (NB * SEQ) / 128), 256, 0, stream>>>(
        Ab, w_proj, b_proj, out);
}
